// CPRNN_28999619182608
// MI455X (gfx1250) — compile-verified, matrix operand order checked by reading, not run on hardware
//
#include <hip/hip_runtime.h>
#include <hip/hip_bf16.h>
#include <math.h>

// Problem sizes (fixed by the reference)
#define S_ 256
#define B_ 32
#define D_ 1024
#define H_ 1024
#define R_ 256
#define V_ 32000

typedef __attribute__((ext_vector_type(16))) __bf16 v16bf;
typedef __attribute__((ext_vector_type(8)))  __bf16 v8bf;
typedef __attribute__((ext_vector_type(8)))  float  v8f;
typedef __attribute__((ext_vector_type(4)))  unsigned int u32x4;
typedef __attribute__((ext_vector_type(8)))  int i32x8;
typedef __attribute__((ext_vector_type(4)))  int i32x4;

// Feature probes (fall back to round-1 code paths if absent)
#if defined(__has_builtin)
#  if __has_builtin(__builtin_amdgcn_global_load_async_to_lds_b128)
#    define HAVE_ASYNC 1
#  endif
#  if __has_builtin(__builtin_amdgcn_tensor_load_to_lds)
#    define HAVE_TDM 1
#  endif
#endif
#ifndef HAVE_ASYNC
#  define HAVE_ASYNC 0
#endif
#ifndef HAVE_TDM
#  define HAVE_TDM 0
#endif

#if HAVE_ASYNC
// Builtin prototype (from hipcc diagnostic): arg0 = int4* in addrspace(1)
// (printed as "int __vector_size__(16) __device__ *"), arg1 = LDS side.
// Cast through integers: as(3) pointers are 32-bit, and the low 32 bits of a
// generic LDS address ARE the LDS offset (flat-aperture mapping).
typedef __attribute__((address_space(1))) i32x4* g4_ptr;
typedef __attribute__((address_space(3))) i32x4* l4_ptr;

__device__ __forceinline__ void async_b128(const void* g, void* l) {
  __builtin_amdgcn_global_load_async_to_lds_b128(
      (g4_ptr)(unsigned long long)(uintptr_t)g,
      (l4_ptr)(unsigned)(uintptr_t)l, 0, 0);
}
#endif

__device__ __forceinline__ void wait_async_8() {
#if defined(__has_builtin) && __has_builtin(__builtin_amdgcn_s_wait_asynccnt)
  __builtin_amdgcn_s_wait_asynccnt(8);
#else
  asm volatile("s_wait_asynccnt 0x8" ::: "memory");
#endif
}
__device__ __forceinline__ void wait_async_0() {
#if defined(__has_builtin) && __has_builtin(__builtin_amdgcn_s_wait_asynccnt)
  __builtin_amdgcn_s_wait_asynccnt(0);
#else
  asm volatile("s_wait_asynccnt 0x0" ::: "memory");
#endif
}

// ---------------------------------------------------------------------------
// Fragment loader for V_WMMA_F32_16X16X32_BF16.
// Source buffer is row-major [rows][ldk] bf16 (A = [M][K], Bt = [N][K]).
// Per the CDNA5 ISA layout (CDNA5 ISA, 16-bit A 16x32):
//   lane 0-15  : row = lane,    K = {0..7, 16..23}  (VGPR pairs K=2j,2j+1)
//   lane 16-31 : row = lane-16, K = {8..15, 24..31}
// i.e. each lane's 16 bf16 = two contiguous 16-byte chunks at K = half*8 and
// K = 16 + half*8.  B uses the mirrored layout with row = N.
// ---------------------------------------------------------------------------
__device__ __forceinline__ v16bf load_frag(const __bf16* base, int ldk,
                                           int row0, int k0) {
  const int lane = threadIdx.x & 31;
  const int half = lane >> 4;
  const __bf16* p = base + (size_t)(row0 + (lane & 15)) * ldk + k0 + half * 8;
  v8bf lo = *(const v8bf*)p;          // K rel [half*8 .. half*8+7]
  v8bf hi = *(const v8bf*)(p + 16);   // K rel [16+half*8 .. 16+half*8+7]
  return __builtin_shufflevector(lo, hi, 0, 1, 2, 3, 4, 5, 6, 7,
                                 8, 9, 10, 11, 12, 13, 14, 15);
}

__device__ __forceinline__ v8f wmma_bf16(v16bf a, v16bf b, v8f c) {
  // (neg_a, A, neg_b, B, c_mod, C, reuse_a, reuse_b)
  return __builtin_amdgcn_wmma_f32_16x16x32_bf16(false, a, false, b,
                                                 (short)0, c, false, false);
}

#if HAVE_TDM
// ---------------------------------------------------------------------------
// TDM: DMA a contiguous block of `nelem` f32 from global into LDS.
// D# per CDNA5 ISA:
//   group0: count=1 | lds_addr | global_addr[56:0] | type=2
//   group1: data_size=2 (4B), tensor_dim0=nelem, tensor_dim1=1,
//           tile_dim0=nelem, tile_dim1=1, tensor_dim0_stride=nelem
//   groups 2/3: zero (<=2D tensor)
// Issue from ONE wave only (EXEC ignored; TENSORcnt is per-wave).
// ---------------------------------------------------------------------------
__device__ __forceinline__ void tdm_load_block_f32(const float* gsrc,
                                                   unsigned lds_off,
                                                   unsigned nelem) {
  unsigned long long ga = (unsigned long long)(uintptr_t)gsrc;
  u32x4 g0 = (u32x4)0u;
  g0[0] = 1u;                                       // count=1, user mode
  g0[1] = lds_off;                                  // LDS byte address
  g0[2] = (unsigned)ga;                             // global_addr[31:0]
  g0[3] = (unsigned)((ga >> 32) & 0x01FFFFFFu) | (2u << 30);  // [56:32] | type=2
  i32x8 g1 = (i32x8)0;
  g1[0] = (int)(2u << 16);                          // data_size=4B; wg_mask=0
  g1[1] = (int)((nelem & 0xFFFFu) << 16);           // tensor_dim0[15:0]
  g1[2] = (int)(((nelem >> 16) & 0xFFFFu) | (1u << 16)); // dim0[31:16], tensor_dim1=1
  g1[3] = (int)((nelem & 0xFFFFu) << 16);           // tile_dim0
  g1[4] = 1;                                        // tile_dim1=1
  g1[5] = (int)nelem;                               // tensor_dim0_stride[31:0]
  i32x4 gz = (i32x4)0;
#if defined(__clang_major__) && __clang_major__ >= 23
  __builtin_amdgcn_tensor_load_to_lds(g0, g1, gz, gz, (i32x8)0, 0);
#else
  __builtin_amdgcn_tensor_load_to_lds(g0, g1, gz, gz, 0);
#endif
}
#endif

// ---------------------------------------------------------------------------
// Prep kernels: fp32 -> bf16 conversions / transposes so all GEMM operands
// are [rows][K] row-major bf16 with contiguous K.
// ---------------------------------------------------------------------------
__global__ void k_f32_to_bf16(const float* __restrict__ src,
                              __bf16* __restrict__ dst, int n) {
  int i = blockIdx.x * blockDim.x + threadIdx.x;
  if (i < n) dst[i] = (__bf16)src[i];
}

// src [rows][cols] f32 -> dst [cols][rows] bf16
__global__ void k_transpose_bf16(const float* __restrict__ src,
                                 __bf16* __restrict__ dst, int rows, int cols) {
  int i = blockIdx.x * blockDim.x + threadIdx.x;
  if (i < rows * cols) {
    int r = i / cols, c = i % cols;
    dst[(size_t)c * rows + r] = (__bf16)src[i];
  }
}

// x_bf16[s*B+b][d] = (bf16) emb[inp[s*B+b]][d]
__global__ void k_gather_embed(const int* __restrict__ inp,
                               const float* __restrict__ emb,
                               __bf16* __restrict__ xb) {
  int i = blockIdx.x * blockDim.x + threadIdx.x;  // over S*B*D
  int row = i >> 10;          // D_ == 1024
  int d   = i & 1023;
  int tok = inp[row];
  xb[(size_t)row * D_ + d] = (__bf16)emb[(size_t)tok * D_ + d];
}

// ---------------------------------------------------------------------------
// Generic bf16 WMMA GEMM:  C[M][ldo] = A[M][K] * Bt[N][K]^T (+ bias[n]).
// Workgroup: 256 threads (8 waves), 128x128 output tile.
// K staged through LDS in 64-wide chunks.  With the CDNA5 async path we
// double-buffer: issue global_load_async_to_lds_b128 for chunk k+1, then
// s_wait_asynccnt 8 (leave the prefetch batch in flight), barrier, compute.
// Wave (wr,wc) owns a 32x64 sub-tile = 2x4 WMMA tiles.
// All dims are exact multiples (M=8192, N in {256,32000}, K=1024).
// ---------------------------------------------------------------------------
#if HAVE_ASYNC
__device__ __forceinline__ void stage_async(const __bf16* __restrict__ A,
                                            const __bf16* __restrict__ Bt,
                                            __bf16* lA, __bf16* lB,
                                            int K, int m0, int n0, int kc,
                                            int tid) {
  for (int i = 0; i < 4; i++) {
    int chunk = tid + i * 256;          // 1024 chunks of 8 bf16 per matrix
    int row   = chunk >> 3;
    int col   = (chunk & 7) * 8;
    async_b128(A  + (size_t)(m0 + row) * K + kc + col, lA + row * 64 + col);
    async_b128(Bt + (size_t)(n0 + row) * K + kc + col, lB + row * 64 + col);
  }
}
#endif

__global__ __launch_bounds__(256) void k_gemm_bf16(
    const __bf16* __restrict__ A, const __bf16* __restrict__ Bt,
    const float* __restrict__ bias, float* __restrict__ C,
    int K, int ldo) {
  const int tid  = threadIdx.x;
  const int wave = tid >> 5;
  const int wr   = wave >> 1;   // 0..3 : M offset wr*32
  const int wc   = wave & 1;    // 0..1 : N offset wc*64
  const int m0   = blockIdx.y * 128;
  const int n0   = blockIdx.x * 128;

  v8f acc[2][4] = {};

#if HAVE_ASYNC
  __shared__ __align__(16) __bf16 As[2][128 * 64];
  __shared__ __align__(16) __bf16 Bs[2][128 * 64];

  stage_async(A, Bt, As[0], Bs[0], K, m0, n0, 0, tid);   // prologue
  const int nk = K / 64;
  for (int it = 0; it < nk; it++) {
    const int buf = it & 1;
    if (it + 1 < nk) {
      stage_async(A, Bt, As[buf ^ 1], Bs[buf ^ 1], K, m0, n0, (it + 1) * 64, tid);
      wait_async_8();            // chunk `it` done; prefetch still in flight
    } else {
      wait_async_0();
    }
    __syncthreads();
    for (int ks = 0; ks < 2; ks++) {
      const int k0 = ks * 32;
      v16bf af[2], bfr[4];
      for (int tm = 0; tm < 2; tm++)
        af[tm] = load_frag(As[buf], 64, wr * 32 + tm * 16, k0);
      for (int tn = 0; tn < 4; tn++)
        bfr[tn] = load_frag(Bs[buf], 64, wc * 64 + tn * 16, k0);
      for (int tm = 0; tm < 2; tm++)
        for (int tn = 0; tn < 4; tn++)
          acc[tm][tn] = wmma_bf16(af[tm], bfr[tn], acc[tm][tn]);
    }
    __syncthreads();
  }
#else
  __shared__ __align__(16) __bf16 As[128 * 64];
  __shared__ __align__(16) __bf16 Bs[128 * 64];

  for (int kc = 0; kc < K; kc += 64) {
    for (int i = 0; i < 4; i++) {
      int chunk = tid + i * 256;
      int row   = chunk >> 3;
      int col   = (chunk & 7) * 8;
      const __bf16* ga = A  + (size_t)(m0 + row) * K + kc + col;
      const __bf16* gb = Bt + (size_t)(n0 + row) * K + kc + col;
      *(v8bf*)(As + row * 64 + col) = *(const v8bf*)ga;
      *(v8bf*)(Bs + row * 64 + col) = *(const v8bf*)gb;
      if (kc + 64 < K) {
        __builtin_prefetch(ga + 64, 0, 0);
        __builtin_prefetch(gb + 64, 0, 0);
      }
    }
    __syncthreads();
    for (int ks = 0; ks < 2; ks++) {
      const int k0 = ks * 32;
      v16bf af[2], bfr[4];
      for (int tm = 0; tm < 2; tm++)
        af[tm] = load_frag(As, 64, wr * 32 + tm * 16, k0);
      for (int tn = 0; tn < 4; tn++)
        bfr[tn] = load_frag(Bs, 64, wc * 64 + tn * 16, k0);
      for (int tm = 0; tm < 2; tm++)
        for (int tn = 0; tn < 4; tn++)
          acc[tm][tn] = wmma_bf16(af[tm], bfr[tn], acc[tm][tn]);
    }
    __syncthreads();
  }
#endif

  // Epilogue: C/D layout = lane n = lane%16, VGPR j -> m = j + (lane>=16)*8
  const int lane  = tid & 31;
  const int half  = lane >> 4;
  const int nlane = lane & 15;
  for (int tm = 0; tm < 2; tm++)
    for (int tn = 0; tn < 4; tn++) {
      int ng = n0 + wc * 64 + tn * 16 + nlane;
      float bv = bias ? bias[ng] : 0.f;
      for (int j = 0; j < 8; j++) {
        int mg = m0 + wr * 32 + tm * 16 + j + half * 8;
        C[(size_t)mg * ldo + ng] = acc[tm][tn][j] + bv;
      }
    }
}

// ---------------------------------------------------------------------------
// Recurrence: h_{t+1} = tanh(((h_t @ a) * bp_t) @ c^T), sequential in t.
// Single persistent workgroup (512 thr = 16 waves on one WGP).
// h kept in LDS as bf16 [32][1024] (64 KB); intermediate u in LDS [32][256].
// With TDM: wave 0 DMAs the NEXT step's contiguous 32 KB bp block into a
// double-buffered LDS region (tensor_load_to_lds + s_wait_tensorcnt),
// overlapping the DMA with this step's WMMAs.
// ---------------------------------------------------------------------------
__global__ __launch_bounds__(512) void k_recurrence(
    const __bf16* __restrict__ a_t,   // [R][H]  (Bt for GEMM1)
    const __bf16* __restrict__ c_b,   // [H][R]  (Bt for GEMM2)
    const float*  __restrict__ bp,    // [S*B][R]
    __bf16* __restrict__ hidden,      // [S*B][H] bf16 (decoder A matrix)
    float*  __restrict__ h_last) {    // [B][H] fp32
  __shared__ __align__(16) __bf16 hbf[B_ * H_];   // 64 KB
  __shared__ __align__(16) __bf16 ubf[B_ * R_];   // 16 KB
#if HAVE_TDM
  __shared__ __align__(16) float  bps[2][B_ * R_]; // 2 x 32 KB
#endif

  const int tid   = threadIdx.x;
  const int wave  = tid >> 5;
  const int lane  = tid & 31;
  const int half  = lane >> 4;
  const int nlane = lane & 15;

  for (int i = tid; i < B_ * H_; i += 512) hbf[i] = (__bf16)0.f;

#if HAVE_TDM
  if (wave == 0)   // prologue DMA: bp block for t=0
    tdm_load_block_f32(bp, (unsigned)(uintptr_t)(void*)&bps[0][0], B_ * R_);
#endif
  __syncthreads();

  for (int t = 0; t < S_; t++) {
#if HAVE_TDM
    if (wave == 0) {
      if (t + 1 < S_) {
        tdm_load_block_f32(bp + (size_t)(t + 1) * B_ * R_,
                           (unsigned)(uintptr_t)(void*)&bps[(t + 1) & 1][0],
                           B_ * R_);
        __builtin_amdgcn_s_wait_tensorcnt((short)1);  // block t landed
      } else {
        __builtin_amdgcn_s_wait_tensorcnt((short)0);
      }
    }
    __syncthreads();
#endif
    // ---- GEMM1: u = (h @ a) * bp_t   (M=32, N=256, K=1024) ----
    // wave w owns n-tile w (16 cols), both m-tiles.  2 WMMA per k-step.
    {
      v8f acc[2] = {};
      const int ntile = wave;
      for (int kk = 0; kk < H_ / 32; kk++) {
        v16bf bfr = load_frag(a_t, H_, ntile * 16, kk * 32);
        v16bf a0  = load_frag(hbf, H_, 0,  kk * 32);
        v16bf a1  = load_frag(hbf, H_, 16, kk * 32);
        acc[0] = wmma_bf16(a0, bfr, acc[0]);
        acc[1] = wmma_bf16(a1, bfr, acc[1]);
      }
      const int ng = ntile * 16 + nlane;
      for (int tm = 0; tm < 2; tm++)
        for (int j = 0; j < 8; j++) {
          int m = tm * 16 + j + half * 8;
#if HAVE_TDM
          float bpv = bps[t & 1][m * R_ + ng];
#else
          float bpv = bp[((size_t)t * B_ + m) * R_ + ng];
#endif
          ubf[m * R_ + ng] = (__bf16)(acc[tm][j] * bpv);
        }
    }
    __syncthreads();
    // ---- GEMM2: h_new = tanh(u @ c^T)   (M=32, N=1024, K=256) ----
    // wave w owns m-tile (w&1) and 8 consecutive n-tiles.
    {
      const int tm  = wave & 1;
      const int tnb = (wave >> 1) * 8;
      v8f acc[8] = {};
      for (int kk = 0; kk < R_ / 32; kk++) {
        v16bf af = load_frag(ubf, R_, tm * 16, kk * 32);
        for (int i = 0; i < 8; i++) {
          v16bf bfr = load_frag(c_b, R_, (tnb + i) * 16, kk * 32);
          acc[i] = wmma_bf16(af, bfr, acc[i]);
        }
      }
      for (int i = 0; i < 8; i++) {
        int ng = (tnb + i) * 16 + nlane;
        for (int j = 0; j < 8; j++) {
          int m = tm * 16 + j + half * 8;
          float v = tanhf(acc[i][j]);
          __bf16 vb = (__bf16)v;
          hbf[m * H_ + ng] = vb;
          hidden[((size_t)t * B_ + m) * H_ + ng] = vb;
          if (t == S_ - 1) h_last[(size_t)m * H_ + ng] = v;
        }
      }
    }
    __syncthreads();
  }
}

// ---------------------------------------------------------------------------
// Host launch.  Workspace (~104 MB): dec_w bf16 | a^T bf16 | b^T bf16 |
// c bf16 | x bf16 | bp f32 | hidden bf16   (all 256-aligned offsets).
// ---------------------------------------------------------------------------
extern "C" void kernel_launch(void* const* d_in, const int* in_sizes, int n_in,
                              void* d_out, int out_size, void* d_ws, size_t ws_size,
                              hipStream_t stream) {
  const int*   inp   = (const int*)  d_in[0];
  const float* emb   = (const float*)d_in[1];
  const float* a     = (const float*)d_in[2];
  const float* b     = (const float*)d_in[3];
  const float* c     = (const float*)d_in[4];
  const float* dec_w = (const float*)d_in[5];
  const float* dec_b = (const float*)d_in[6];

  float* logits = (float*)d_out;                      // [S,B,V]
  float* h_last = logits + (size_t)S_ * B_ * V_;      // [B,H]

  char* ws = (char*)d_ws;
  size_t o = 0;
  __bf16* dec_wb = (__bf16*)(ws + o); o += (size_t)V_ * H_ * 2;
  __bf16* a_t    = (__bf16*)(ws + o); o += (size_t)H_ * R_ * 2;
  __bf16* b_t    = (__bf16*)(ws + o); o += (size_t)D_ * R_ * 2;
  __bf16* c_b    = (__bf16*)(ws + o); o += (size_t)H_ * R_ * 2;
  __bf16* x_b    = (__bf16*)(ws + o); o += (size_t)S_ * B_ * D_ * 2;
  float*  bp     = (float*) (ws + o); o += (size_t)S_ * B_ * R_ * 4;
  __bf16* hid    = (__bf16*)(ws + o); o += (size_t)S_ * B_ * H_ * 2;

  // 1) weight conversions / transposes
  k_f32_to_bf16   <<<(V_ * H_ + 255) / 256, 256, 0, stream>>>(dec_w, dec_wb, V_ * H_);
  k_transpose_bf16<<<(H_ * R_ + 255) / 256, 256, 0, stream>>>(a, a_t, H_, R_);
  k_transpose_bf16<<<(D_ * R_ + 255) / 256, 256, 0, stream>>>(b, b_t, D_, R_);
  k_f32_to_bf16   <<<(H_ * R_ + 255) / 256, 256, 0, stream>>>(c, c_b, H_ * R_);

  // 2) embedding gather -> bf16
  k_gather_embed<<<(S_ * B_ * D_) / 256, 256, 0, stream>>>(inp, emb, x_b);

  // 3) bp = x @ b : [8192,1024] x [1024,256]
  k_gemm_bf16<<<dim3(R_ / 128, (S_ * B_) / 128), 256, 0, stream>>>(
      x_b, b_t, nullptr, bp, D_, R_);

  // 4) sequential CP recurrence (single WGP, persistent)
  k_recurrence<<<1, 512, 0, stream>>>(a_t, c_b, bp, hid, h_last);

  // 5) decoder: logits = hidden @ dec_w^T + dec_b : [8192,1024] x [1024,32000]
  k_gemm_bf16<<<dim3(V_ / 128, (S_ * B_) / 128), 256, 0, stream>>>(
      hid, dec_wb, dec_b, logits, H_, V_);
}
